// Head_48515950576006
// MI455X (gfx1250) — matrix operand order repaired and checked by reading, not compiled, not run on hardware
//
#include <hip/hip_runtime.h>

typedef __attribute__((ext_vector_type(16))) _Float16 v16h;
typedef __attribute__((ext_vector_type(8)))  _Float16 v8h;
typedef __attribute__((ext_vector_type(8)))  float    v8f;
typedef __attribute__((ext_vector_type(4)))  unsigned int u32x4;
typedef __attribute__((ext_vector_type(8)))  int          i32x8;
typedef __attribute__((ext_vector_type(4)))  int          i32x4;

#define B_  8
#define T_  2048
#define C_  1024
#define H_  64
#define NROW (B_*T_)

// workspace layout (bytes)
#define WP_OFF ((size_t)0)                       // packed W: 3*1024*64 f16 = 384KB
#define Q_OFF  ((size_t)(512*1024))              // q f16 [row][h]  : 2MB
#define K_OFF  (Q_OFF + (size_t)2*1024*1024)     // k f16 [row][h]  : 2MB
#define V_OFF  (K_OFF + (size_t)2*1024*1024)     // v f16 [b][h][t] : 2MB

// --------------------------------------------------------------------------
// Pack Wq/Wk/Wv (f32 [C][H]) into f16 B-fragment order:
// frag index = ((mat*32 + kc)*4 + nt)*32 + lane, 16 halves per entry.
// B 32x16 layout: lane -> N = lane%16; lanes 0-15 hold K=0..7,16..23, 16-31 hold K=8..15,24..31.
// --------------------------------------------------------------------------
__global__ void pack_w_kernel(const float* __restrict__ Wq,
                              const float* __restrict__ Wk,
                              const float* __restrict__ Wv,
                              _Float16* __restrict__ wp)
{
    int idx = blockIdx.x * blockDim.x + threadIdx.x;
    if (idx >= 3*32*4*32) return;
    int lane = idx & 31;
    int nt   = (idx >> 5) & 3;
    int kc   = (idx >> 7) & 31;
    int mat  = idx >> 12;
    const float* W = (mat == 0) ? Wq : (mat == 1) ? Wk : Wv;
    int g = lane >> 4, n = lane & 15;
    _Float16* dst = wp + (size_t)idx * 16;
#pragma unroll
    for (int e = 0; e < 16; ++e) {
        int k = kc*32 + (e < 8 ? 8*g + e : 16 + 8*g + (e - 8));
        dst[e] = (_Float16)W[k*H_ + nt*16 + n];
    }
}

// --------------------------------------------------------------------------
// QKV projection, TDM edition. gridDim = (128, 3): blockIdx.y selects the
// matrix (q/k/v). Each block (8 waves, 128 rows) first DMAs the 128KB packed
// weight matrix into LDS with the Tensor Data Mover, then all 8 waves feed
// their WMMAs from ds_load_b128 instead of VMEM.
// --------------------------------------------------------------------------
__global__ __launch_bounds__(256) void qkv_kernel(const float*    __restrict__ x,
                                                  const _Float16* __restrict__ wp,
                                                  _Float16* __restrict__ qb,
                                                  _Float16* __restrict__ kb,
                                                  _Float16* __restrict__ vt)
{
    extern __shared__ _Float16 sw[];                   // 128KB dynamic LDS
    int mat  = blockIdx.y;
    int wid  = threadIdx.x >> 5;
    int lane = threadIdx.x & 31;
    int g = lane >> 4, n = lane & 15;
    int row0 = blockIdx.x * 128 + wid * 16;

    // ---- TDM: one wave DMAs packed W[mat] (1024x64 f16 = 128KB) into LDS.
    if (wid == 0) {
        unsigned long long ga = (unsigned long long)(uintptr_t)(wp + (size_t)mat * C_ * H_);
        unsigned int lds = (unsigned int)(uintptr_t)(void*)sw;   // LDS byte offset
        const unsigned int dim0 = (C_ * H_ * 2) / 8;             // 16384 8-byte elems

        u32x4 g0;
        g0[0] = 1u;                                    // count=1, user mode
        g0[1] = lds;                                   // lds_addr
        g0[2] = (unsigned int)ga;                      // global_addr[31:0]
        g0[3] = (unsigned int)(ga >> 32) | (2u << 30); // global_addr[56:32] | type=2

        i32x8 g1;
        g1[0] = (int)(3u << 16);                       // data_size = 8B, mask=0
        g1[1] = (int)((dim0 & 0xFFFFu) << 16);         // tensor_dim0[15:0]
        g1[2] = (int)((dim0 >> 16) | (1u << 16));      // tensor_dim0[31:16] | tensor_dim1=1
        g1[3] = (int)((dim0 & 0xFFFFu) << 16);         // tile_dim0 = dim0
        g1[4] = 1;                                     // tile_dim1 = 1, tile_dim2 = 0
        g1[5] = (int)dim0;                             // tensor_dim0_stride[31:0]
        g1[6] = 0;                                     // stride0[47:32] | stride1[15:0]
        g1[7] = 0;

        i32x4 z4;
        z4[0] = 0; z4[1] = 0; z4[2] = 0; z4[3] = 0;    // 2D tensor: groups 2/3 unused
        i32x8 z8;
#pragma unroll
        for (int i = 0; i < 8; ++i) z8[i] = 0;
        __builtin_amdgcn_tensor_load_to_lds(g0, g1, z4, z4, z8, 0);
        __builtin_amdgcn_s_wait_tensorcnt(0);
    }
    __syncthreads();

    // ---- GEMM: 16 rows x 64 cols per wave, K = 1024 in 32 chunks.
    const float* xrow = x + (size_t)(row0 + n) * C_;   // A-frag: M = lane%16
    const v16h* swf = (const v16h*)sw;

    v8f acc[4];
#pragma unroll
    for (int t = 0; t < 4; ++t)
#pragma unroll
        for (int r = 0; r < 8; ++r) acc[t][r] = 0.0f;

    for (int kc = 0; kc < 32; ++kc) {
        // A-frag 16x32 f16: lanes 0-15 K{0-7,16-23}, lanes 16-31 K{8-15,24-31}
        int kbase = kc*32 + 8*g;
        __builtin_prefetch((const void*)(xrow + kbase + 32), 0, 3); // next chunk
        float4 f0 = *(const float4*)(xrow + kbase);
        float4 f1 = *(const float4*)(xrow + kbase + 4);
        float4 f2 = *(const float4*)(xrow + kbase + 16);
        float4 f3 = *(const float4*)(xrow + kbase + 20);
        float fv[16] = { f0.x,f0.y,f0.z,f0.w, f1.x,f1.y,f1.z,f1.w,
                         f2.x,f2.y,f2.z,f2.w, f3.x,f3.y,f3.z,f3.w };
        v16h a;
#pragma unroll
        for (int e = 0; e < 16; ++e) a[e] = (_Float16)fv[e];

#pragma unroll
        for (int t = 0; t < 4; ++t) {
            v16h bf = swf[(kc*4 + t)*32 + lane];       // ds_load_b128 x2
            acc[t] = __builtin_amdgcn_wmma_f32_16x16x32_f16(
                false, a, false, bf, (short)0, acc[t], false, false);
        }
    }

    // ---- epilogue; C-frag: M = r + 8*(lane/16), N = lane%16
#pragma unroll
    for (int t = 0; t < 4; ++t) {
#pragma unroll
        for (int r = 0; r < 8; ++r) {
            int row = row0 + r + 8*g;
            int h   = t*16 + n;
            size_t qi = (size_t)row * H_ + h;
            if (mat == 0) {
                qb[qi] = (_Float16)(acc[t][r] * 0.125f);        // fold 1/sqrt(H)
            } else if (mat == 1) {
                kb[qi] = (_Float16)acc[t][r];
            } else {
                int bb = row >> 11, tl = row & (T_-1);
                vt[((size_t)bb*H_ + h)*T_ + tl] = (_Float16)acc[t][r];
            }
        }
    }
}

// --------------------------------------------------------------------------
// Key-split flash attention: one block = one 16-query tile, 4 waves.
// Wave w processes 32-key blocks s0 = w*32, w*32+128, ... with its own
// online-softmax state; partials merged via LDS (log-sum-exp).
// grid = B * T/16 = 1024 blocks -> 4096 waves.
// --------------------------------------------------------------------------
__global__ __launch_bounds__(128) void attn_kernel(const _Float16* __restrict__ qb,
                                                   const _Float16* __restrict__ kb,
                                                   const _Float16* __restrict__ vt,
                                                   float* __restrict__ out)
{
    __shared__ _Float16 pbuf[4][16*32];     // per-wave P staging
    __shared__ float cm[4][16];             // per-wave row max
    __shared__ float cl[4][16];             // per-wave row sum
    __shared__ float co[4][4][8][32];       // per-wave O partials [w][nt][r][lane]

    int wid  = threadIdx.x >> 5;
    int lane = threadIdx.x & 31;
    int g = lane >> 4, n = lane & 15;
    int b  = blockIdx.x >> 7;               // batch
    int q0 = (blockIdx.x & 127) << 4;       // query tile start

    // Q A-frags for K-dim (h) = 0..31, 32..63 (identical across waves; L2 hit)
    const _Float16* qrow = qb + ((size_t)b*T_ + q0 + n) * H_;
    v16h qa[2];
#pragma unroll
    for (int ks = 0; ks < 2; ++ks) {
        v8h lo = *(const v8h*)(qrow + ks*32 + 16*g);
        v8h hi = *(const v8h*)(qrow + ks*32 + 16*g + 8);
#pragma unroll
        for (int e = 0; e < 8; ++e) { qa[ks][e] = lo[e]; qa[ks][e+8] = hi[e]; }
    }

    float mrow[8], lrow[8];
    v8f o[4];
#pragma unroll
    for (int r = 0; r < 8; ++r) { mrow[r] = -1e30f; lrow[r] = 0.0f; }
#pragma unroll
    for (int t = 0; t < 4; ++t)
#pragma unroll
        for (int r = 0; r < 8; ++r) o[t][r] = 0.0f;

    v8f zacc;
#pragma unroll
    for (int r = 0; r < 8; ++r) zacc[r] = 0.0f;

    _Float16* pw = &pbuf[wid][0];
    int kend = q0 + 16;                     // exclusive end of valid keys
    for (int s0 = wid*32; s0 < kend; s0 += 128) {
        // ---- S = (Q*scale) . K^T for two 16-key tiles
        v8f s[2];
#pragma unroll
        for (int j = 0; j < 2; ++j) {
            const _Float16* krow = kb + ((size_t)b*T_ + s0 + 16*j + n) * H_;
            v16h kf0 = *(const v16h*)(krow + 16*g);        // h = 0..31
            v16h kf1 = *(const v16h*)(krow + 32 + 16*g);   // h = 32..63
            s[j] = __builtin_amdgcn_wmma_f32_16x16x32_f16(false, qa[0], false, kf0, (short)0, zacc, false, false);
            s[j] = __builtin_amdgcn_wmma_f32_16x16x32_f16(false, qa[1], false, kf1, (short)0, s[j], false, false);
        }

        // ---- issue V-fragment loads NOW; latency hides under the softmax
        v16h vf[4];
#pragma unroll
        for (int t = 0; t < 4; ++t) {
            const _Float16* vrow = vt + ((size_t)b*H_ + t*16 + n) * T_ + s0 + 16*g;
            vf[t] = *(const v16h*)vrow;
        }

        // ---- causal mask (boundary blocks only)
        if (s0 + 31 > q0) {
#pragma unroll
            for (int j = 0; j < 2; ++j) {
                int key = s0 + 16*j + n;
#pragma unroll
                for (int r = 0; r < 8; ++r) {
                    int qi = q0 + r + 8*g;
                    if (key > qi) s[j][r] = -1e30f;
                }
            }
        }

        // ---- online softmax; rows striped across 16-lane groups
        float p0[8], p1[8];
#pragma unroll
        for (int r = 0; r < 8; ++r) {
            float v = fmaxf(s[0][r], s[1][r]);
#pragma unroll
            for (int msk = 8; msk >= 1; msk >>= 1)
                v = fmaxf(v, __shfl_xor(v, msk, 32));
            float mn = fmaxf(mrow[r], v);
            float c  = __expf(mrow[r] - mn);
            p0[r] = __expf(s[0][r] - mn);
            p1[r] = __expf(s[1][r] - mn);
            float rs = p0[r] + p1[r];
#pragma unroll
            for (int msk = 8; msk >= 1; msk >>= 1)
                rs += __shfl_xor(rs, msk, 32);
            lrow[r] = lrow[r]*c + rs;
            mrow[r] = mn;
#pragma unroll
            for (int t = 0; t < 4; ++t) o[t][r] *= c;
        }

        // ---- P: C-layout -> LDS -> A-frag (f16)
#pragma unroll
        for (int r = 0; r < 8; ++r) {
            int M = r + 8*g;
            pw[M*32 + n]      = (_Float16)p0[r];
            pw[M*32 + 16 + n] = (_Float16)p1[r];
        }
        asm volatile("s_wait_dscnt 0" ::: "memory");
        v16h pa;
        {
            v8h lo = *(const v8h*)(pw + n*32 + 16*g);       // K = 16g..16g+7
            v8h hi = *(const v8h*)(pw + n*32 + 16*g + 8);   // K = 16g+8..16g+15
#pragma unroll
            for (int e = 0; e < 8; ++e) { pa[e] = lo[e]; pa[e+8] = hi[e]; }
        }

        // ---- O += P . V
#pragma unroll
        for (int t = 0; t < 4; ++t)
            o[t] = __builtin_amdgcn_wmma_f32_16x16x32_f16(
                false, pa, false, vf[t], (short)0, o[t], false, false);
    }

    // ---- publish per-wave partial state
    if (n == 0) {
#pragma unroll
        for (int r = 0; r < 8; ++r) {
            cm[wid][r + 8*g] = mrow[r];
            cl[wid][r + 8*g] = lrow[r];
        }
    }
#pragma unroll
    for (int t = 0; t < 4; ++t)
#pragma unroll
        for (int r = 0; r < 8; ++r)
            co[wid][t][r][lane] = o[t][r];
    __syncthreads();

    // ---- wave 0 merges the 4 partials (log-sum-exp), stores fp32 out[b][t][h]
    if (wid == 0) {
#pragma unroll
        for (int r = 0; r < 8; ++r) {
            int M = r + 8*g;
            float mm = cm[0][M];
#pragma unroll
            for (int w = 1; w < 4; ++w) mm = fmaxf(mm, cm[w][M]);
            float wgt[4];
            float L = 0.0f;
#pragma unroll
            for (int w = 0; w < 4; ++w) {
                wgt[w] = __expf(cm[w][M] - mm);
                L += cl[w][M] * wgt[w];
            }
            float inv = 1.0f / L;
#pragma unroll
            for (int t = 0; t < 4; ++t) {
                float acc = 0.0f;
#pragma unroll
                for (int w = 0; w < 4; ++w) acc += co[w][t][r][lane] * wgt[w];
                out[((size_t)b*T_ + q0 + M)*H_ + t*16 + n] = acc * inv;
            }
        }
    }
}

// --------------------------------------------------------------------------
extern "C" void kernel_launch(void* const* d_in, const int* in_sizes, int n_in,
                              void* d_out, int out_size, void* d_ws, size_t ws_size,
                              hipStream_t stream)
{
    const float* x  = (const float*)d_in[0];
    const float* Wq = (const float*)d_in[1];
    const float* Wk = (const float*)d_in[2];
    const float* Wv = (const float*)d_in[3];
    float* out = (float*)d_out;

    char* ws = (char*)d_ws;
    _Float16* wp = (_Float16*)(ws + WP_OFF);
    _Float16* qb = (_Float16*)(ws + Q_OFF);
    _Float16* kb = (_Float16*)(ws + K_OFF);
    _Float16* vt = (_Float16*)(ws + V_OFF);

    hipLaunchKernelGGL(pack_w_kernel, dim3(48), dim3(256), 0, stream, Wq, Wk, Wv, wp);
    hipLaunchKernelGGL(qkv_kernel, dim3(NROW/128, 3), dim3(256),
                       (size_t)C_ * H_ * sizeof(_Float16) /* 128KB LDS */, stream,
                       x, wp, qb, kb, vt);
    hipLaunchKernelGGL(attn_kernel, dim3(B_*(T_/16)), dim3(128), 0, stream, qb, kb, vt, out);
}
